// MPNNModel_8031588844023
// MI455X (gfx1250) — compile-verified, matrix operand order checked by reading, not run on hardware
//
#include <hip/hip_runtime.h>
#include <hip/hip_fp16.h>

typedef __attribute__((ext_vector_type(16))) _Float16 v16h;
typedef __attribute__((ext_vector_type(8)))  float    v8f;

#define NN 50000      // nodes
#define NE 800000     // edges
#define IND 128
#define EMB 32
#define EDD 8
#define NL 4
#define NB 64
#define NOUT 10
#define BN_EPS 1e-5f

// ---------------------------------------------------------------------------
// WMMA fragment helpers (wave32, 16x16x32 f16, f32 accumulate).
// A-matrix 16x32 f16 layout (ISA 7.12.2): lanes 0-15 -> M=lane, K pairs
// {0,1},{2,3},{4,5},{6,7} in VGPR0-3 and {16..23} in VGPR4-7; lanes 16-31 ->
// M=lane-16 with K offset +8. B fragments are pre-packed host-side with the
// mirrored mapping (lane -> N column).
// ---------------------------------------------------------------------------
union AFrag { v16h v; _Float16 h[16]; unsigned u[8]; uint4 q[2]; };

__device__ __forceinline__ int elemk(int j, int hi) {
  int p = j >> 1;
  int base = (p < 4) ? (2 * p) : (16 + 2 * (p - 4));
  return base + (j & 1) + (hi ? 8 : 0);
}

__device__ __forceinline__ v8f wmma_f16(v16h a, v16h b, v8f c) {
  return __builtin_amdgcn_wmma_f32_16x16x32_f16(false, a, false, b, (short)0, c,
                                                false, false);
}

// Load one lane's A-fragment slice from a row-major f16 row (32-wide K chunk):
// exactly two 16B loads (k = hi*8..hi*8+7 and 16+hi*8..16+hi*8+7).
__device__ __forceinline__ void load_a_f16(AFrag& a, const _Float16* row, int hi) {
  a.q[0] = *(const uint4*)(row + (hi ? 8 : 0));
  a.q[1] = *(const uint4*)(row + 16 + (hi ? 8 : 0));
}

// Same but from an f32 row, converting pairs to f16.
__device__ __forceinline__ void load_a_f32(AFrag& a, const float* row, int hi) {
  const float* r0 = row + (hi ? 8 : 0);
  const float* r1 = r0 + 16;
#pragma unroll
  for (int p = 0; p < 4; ++p) {
    float2 x0 = *(const float2*)(r0 + 2 * p);
    float2 x1 = *(const float2*)(r1 + 2 * p);
    a.h[2 * p]     = (_Float16)x0.x;  a.h[2 * p + 1] = (_Float16)x0.y;
    a.h[8 + 2 * p] = (_Float16)x1.x;  a.h[9 + 2 * p] = (_Float16)x1.y;
  }
}

// ---------------------------------------------------------------------------
// Weight packing: [Krows,32] f32 row-major -> per-(Kchunk,Nhalf) B fragments,
// 512 f16 each, [lane][16 halves] order, zero-padded K.
// ---------------------------------------------------------------------------
__global__ void pack_b_frags(const float* __restrict__ Wsrc, int Krows,
                             _Float16* __restrict__ dst) {
  int f = blockIdx.x;
  int kc = f >> 1, nh = f & 1;
  for (int idx = threadIdx.x; idx < 512; idx += blockDim.x) {
    int lane = idx >> 4, j = idx & 15;
    int n = nh * 16 + (lane & 15);
    int k = kc * 32 + elemk(j, lane >> 4);
    float w = (k < Krows) ? Wsrc[k * 32 + n] : 0.f;
    dst[f * 512 + idx] = (_Float16)w;
  }
}

__global__ void cvt_f16(const float* __restrict__ s, _Float16* __restrict__ d) {
  int i = blockIdx.x * blockDim.x + threadIdx.x;
  d[i] = (_Float16)s[i];
}

// ---------------------------------------------------------------------------
// h = x @ W_in + b : [50000,128]@[128,32]. 5 waves/block, one 16-row tile per
// wave, 4 K-chunks x 2 N-halves = 8 WMMA. Writes h (f32) and hf (f16).
// ---------------------------------------------------------------------------
__global__ void lin_in_gemm(const float* __restrict__ x,
                            const _Float16* __restrict__ wfrag,
                            const float* __restrict__ bias,
                            float* __restrict__ h, _Float16* __restrict__ hf) {
  __shared__ float tile[5][512];
  int wave = threadIdx.x >> 5, lane = threadIdx.x & 31;
  int t = blockIdx.x * 5 + wave;          // 625 blocks -> 3125 tiles exactly
  int hi = lane >> 4, m = lane & 15;
  const v16h* B = (const v16h*)wfrag;
  v8f acc0 = {}, acc1 = {};
#pragma unroll
  for (int kc = 0; kc < 4; ++kc) {
    AFrag a;
    load_a_f32(a, x + (size_t)(t * 16 + m) * IND + kc * 32, hi);
    acc0 = wmma_f16(a.v, B[(kc * 2 + 0) * 32 + lane], acc0);
    acc1 = wmma_f16(a.v, B[(kc * 2 + 1) * 32 + lane], acc1);
  }
  float b0 = bias[m], b1 = bias[16 + m];
#pragma unroll
  for (int r = 0; r < 8; ++r) {           // D: lane holds column n, rows r+hi*8
    tile[wave][(r + hi * 8) * 32 + m] = acc0[r] + b0;
    tile[wave][(r + hi * 8) * 32 + 16 + m] = acc1[r] + b1;
  }
  __syncthreads();
#pragma unroll
  for (int i = 0; i < 16; ++i) {
    int idx = lane * 16 + i;
    float v = tile[wave][idx];
    h[(size_t)t * 512 + idx] = v;
    hf[(size_t)t * 512 + idx] = (_Float16)v;
  }
}

// ---------------------------------------------------------------------------
// Message GEMM1: m1 = concat(h[dst],h[src],ea) @ W1 + b (raw, pre-BN).
// Per 16-edge tile: 3 K-chunks x 2 N-halves = 6 WMMA. Accumulates per-channel
// sum/sumsq for BN. Output stored in A-fragment order (via LDS transpose) so
// GEMM2 loads it with one 32B read per lane.
// ---------------------------------------------------------------------------
__global__ void msg_gemm1(const _Float16* __restrict__ hf,
                          const _Float16* __restrict__ eaf,
                          const int* __restrict__ ei,
                          const _Float16* __restrict__ wfrag,
                          const float* __restrict__ bias,
                          _Float16* __restrict__ m1f,
                          float* __restrict__ gSum, float* __restrict__ gSqs) {
  __shared__ _Float16 tile[4][512];
  __shared__ float sSum[32], sSqs[32];
  if (threadIdx.x < 32) { sSum[threadIdx.x] = 0.f; sSqs[threadIdx.x] = 0.f; }
  __syncthreads();
  int wave = threadIdx.x >> 5, lane = threadIdx.x & 31;
  int t = blockIdx.x * 4 + wave;          // 12500 blocks -> 50000 tiles exactly
  int hi = lane >> 4, m = lane & 15;
  int e = t * 16 + m;
  __builtin_prefetch(ei + e + 2048, 0, 1);
  int sIdx = ei[e], dIdx = ei[NE + e];
  AFrag a0, a1, a2;
  load_a_f16(a0, hf + (size_t)dIdx * 32, hi);   // k 0..31  : h[dst]
  load_a_f16(a1, hf + (size_t)sIdx * 32, hi);   // k 32..63 : h[src]
  a2.q[0] = make_uint4(0, 0, 0, 0);             // k 64..95 : ea (8 real, pad 0)
  a2.q[1] = make_uint4(0, 0, 0, 0);
  if (!hi) a2.q[0] = *(const uint4*)(eaf + (size_t)e * 8);
  const v16h* B = (const v16h*)wfrag;
  v8f acc0 = {}, acc1 = {};
  acc0 = wmma_f16(a0.v, B[0 * 32 + lane], acc0);
  acc1 = wmma_f16(a0.v, B[1 * 32 + lane], acc1);
  acc0 = wmma_f16(a1.v, B[2 * 32 + lane], acc0);
  acc1 = wmma_f16(a1.v, B[3 * 32 + lane], acc1);
  acc0 = wmma_f16(a2.v, B[4 * 32 + lane], acc0);
  acc1 = wmma_f16(a2.v, B[5 * 32 + lane], acc1);
  float b0 = bias[m], b1 = bias[16 + m];
  float s0 = 0, q0 = 0, s1 = 0, q1 = 0;
#pragma unroll
  for (int r = 0; r < 8; ++r) {
    float v0 = acc0[r] + b0, v1 = acc1[r] + b1;
    s0 += v0; q0 += v0 * v0; s1 += v1; q1 += v1 * v1;
    tile[wave][(r + hi * 8) * 32 + m] = (_Float16)v0;
    tile[wave][(r + hi * 8) * 32 + 16 + m] = (_Float16)v1;
  }
  atomicAdd(&sSum[m], s0);      atomicAdd(&sSqs[m], q0);
  atomicAdd(&sSum[16 + m], s1); atomicAdd(&sSqs[16 + m], q1);
  __syncthreads();
  if (threadIdx.x < 32) {
    atomicAdd(&gSum[threadIdx.x], sSum[threadIdx.x]);
    atomicAdd(&gSqs[threadIdx.x], sSqs[threadIdx.x]);
  }
  AFrag o;                                  // re-read tile in A-frag order
  load_a_f16(o, &tile[wave][m * 32], hi);
  uint4* dst = (uint4*)(m1f + ((size_t)t * 32 + lane) * 16);
  dst[0] = o.q[0]; dst[1] = o.q[1];
}

// ---------------------------------------------------------------------------
// Generic GEMM2: out = BN(in1_frag)*ReLU @ W2 + b (raw, pre-BN), K=32.
// Used for both message (E tiles, 4 waves/blk) and update (N tiles, 5 w/blk).
// Output stored row-major [rows,32] f16.
// ---------------------------------------------------------------------------
__global__ void gemm2_kernel(const _Float16* __restrict__ inf,
                             const _Float16* __restrict__ wfrag,
                             const float* __restrict__ scale,
                             const float* __restrict__ shift,
                             const float* __restrict__ bias,
                             _Float16* __restrict__ outp,
                             float* __restrict__ gSum, float* __restrict__ gSqs) {
  __shared__ _Float16 tile[5][512];
  __shared__ float sSum[32], sSqs[32];
  if (threadIdx.x < 32) { sSum[threadIdx.x] = 0.f; sSqs[threadIdx.x] = 0.f; }
  __syncthreads();
  int wave = threadIdx.x >> 5, lane = threadIdx.x & 31;
  int t = blockIdx.x * (blockDim.x >> 5) + wave;
  int hi = lane >> 4, m = lane & 15;
  AFrag a;
  const uint4* src = (const uint4*)(inf + ((size_t)t * 32 + lane) * 16);
  a.q[0] = src[0]; a.q[1] = src[1];
#pragma unroll
  for (int j = 0; j < 16; ++j) {            // apply BN + ReLU per element
    int k = elemk(j, hi);
    float v = (float)a.h[j] * scale[k] + shift[k];
    a.h[j] = (_Float16)fmaxf(v, 0.f);
  }
  const v16h* B = (const v16h*)wfrag;
  v8f acc0 = {}, acc1 = {};
  acc0 = wmma_f16(a.v, B[lane], acc0);
  acc1 = wmma_f16(a.v, B[32 + lane], acc1);
  float b0 = bias[m], b1 = bias[16 + m];
  float s0 = 0, q0 = 0, s1 = 0, q1 = 0;
#pragma unroll
  for (int r = 0; r < 8; ++r) {
    float v0 = acc0[r] + b0, v1 = acc1[r] + b1;
    s0 += v0; q0 += v0 * v0; s1 += v1; q1 += v1 * v1;
    tile[wave][(r + hi * 8) * 32 + m] = (_Float16)v0;
    tile[wave][(r + hi * 8) * 32 + 16 + m] = (_Float16)v1;
  }
  atomicAdd(&sSum[m], s0);      atomicAdd(&sSqs[m], q0);
  atomicAdd(&sSum[16 + m], s1); atomicAdd(&sSqs[16 + m], q1);
  __syncthreads();
  if (threadIdx.x < 32) {
    atomicAdd(&gSum[threadIdx.x], sSum[threadIdx.x]);
    atomicAdd(&gSqs[threadIdx.x], sSqs[threadIdx.x]);
  }
  const uint4* s2 = (const uint4*)(&tile[wave][lane * 16]);
  uint4* dst = (uint4*)(outp + (size_t)t * 512 + lane * 16);
  dst[0] = s2[0]; dst[1] = s2[1];
}

// ---------------------------------------------------------------------------
// Update GEMM1: u1 = concat(h, aggr) @ Wu1 + b (raw). K=64 -> 2 chunks.
// ---------------------------------------------------------------------------
__global__ void upd_gemm1(const _Float16* __restrict__ hf,
                          const float* __restrict__ aggr,
                          const _Float16* __restrict__ wfrag,
                          const float* __restrict__ bias,
                          _Float16* __restrict__ u1f,
                          float* __restrict__ gSum, float* __restrict__ gSqs) {
  __shared__ _Float16 tile[5][512];
  __shared__ float sSum[32], sSqs[32];
  if (threadIdx.x < 32) { sSum[threadIdx.x] = 0.f; sSqs[threadIdx.x] = 0.f; }
  __syncthreads();
  int wave = threadIdx.x >> 5, lane = threadIdx.x & 31;
  int t = blockIdx.x * 5 + wave;           // 625 blocks -> 3125 tiles exactly
  int hi = lane >> 4, m = lane & 15;
  int row = t * 16 + m;
  AFrag a0, a1;
  load_a_f16(a0, hf + (size_t)row * 32, hi);
  load_a_f32(a1, aggr + (size_t)row * 32, hi);
  const v16h* B = (const v16h*)wfrag;
  v8f acc0 = {}, acc1 = {};
  acc0 = wmma_f16(a0.v, B[0 * 32 + lane], acc0);
  acc1 = wmma_f16(a0.v, B[1 * 32 + lane], acc1);
  acc0 = wmma_f16(a1.v, B[2 * 32 + lane], acc0);
  acc1 = wmma_f16(a1.v, B[3 * 32 + lane], acc1);
  float b0 = bias[m], b1 = bias[16 + m];
  float s0 = 0, q0 = 0, s1 = 0, q1 = 0;
#pragma unroll
  for (int r = 0; r < 8; ++r) {
    float v0 = acc0[r] + b0, v1 = acc1[r] + b1;
    s0 += v0; q0 += v0 * v0; s1 += v1; q1 += v1 * v1;
    tile[wave][(r + hi * 8) * 32 + m] = (_Float16)v0;
    tile[wave][(r + hi * 8) * 32 + 16 + m] = (_Float16)v1;
  }
  atomicAdd(&sSum[m], s0);      atomicAdd(&sSqs[m], q0);
  atomicAdd(&sSum[16 + m], s1); atomicAdd(&sSqs[16 + m], q1);
  __syncthreads();
  if (threadIdx.x < 32) {
    atomicAdd(&gSum[threadIdx.x], sSum[threadIdx.x]);
    atomicAdd(&gSqs[threadIdx.x], sSqs[threadIdx.x]);
  }
  AFrag o;
  load_a_f16(o, &tile[wave][m * 32], hi);
  uint4* dst = (uint4*)(u1f + ((size_t)t * 32 + lane) * 16);
  dst[0] = o.q[0]; dst[1] = o.q[1];
}

// BN stats -> (scale, shift): scale = g*rsqrt(var+eps), shift = be - mean*scale
__global__ void finalize_stats(const float* __restrict__ gSum,
                               const float* __restrict__ gSqs,
                               const float* __restrict__ g,
                               const float* __restrict__ be, float cnt,
                               float* __restrict__ scale, float* __restrict__ shift) {
  int c = threadIdx.x;
  float mean = gSum[c] / cnt;
  float var = fmaxf(gSqs[c] / cnt - mean * mean, 0.f);
  float s = g[c] * rsqrtf(var + BN_EPS);
  scale[c] = s;
  shift[c] = be[c] - mean * s;
}

// aggr[dst] += relu(bn(m2))  — one f16 pair per thread, f32 L2 atomics.
__global__ void scatter_kernel(const _Float16* __restrict__ m2,
                               const int* __restrict__ ei,
                               const float* __restrict__ scale,
                               const float* __restrict__ shift,
                               float* __restrict__ aggr) {
  int idx = blockIdx.x * blockDim.x + threadIdx.x;   // E*16 threads
  int e = idx >> 4, ch = (idx & 15) * 2;
  union { unsigned u; _Float16 h2[2]; } p;
  p.u = *(const unsigned*)(m2 + (size_t)e * 32 + ch);
  int d = ei[NE + e];
  float a0 = fmaxf((float)p.h2[0] * scale[ch] + shift[ch], 0.f);
  float a1 = fmaxf((float)p.h2[1] * scale[ch + 1] + shift[ch + 1], 0.f);
  atomicAdd(&aggr[(size_t)d * 32 + ch], a0);
  atomicAdd(&aggr[(size_t)d * 32 + ch + 1], a1);
}

// h = relu(bn(u2)) + h  (residual); refresh f16 mirror.
__global__ void resid_kernel(const _Float16* __restrict__ u2,
                             const float* __restrict__ scale,
                             const float* __restrict__ shift,
                             float* __restrict__ h, _Float16* __restrict__ hf) {
  int idx = blockIdx.x * blockDim.x + threadIdx.x;   // N*16 threads
  int v = idx >> 4, ch = (idx & 15) * 2;
  union { unsigned u; _Float16 h2[2]; } p;
  p.u = *(const unsigned*)(u2 + (size_t)v * 32 + ch);
  float a0 = fmaxf((float)p.h2[0] * scale[ch] + shift[ch], 0.f);
  float a1 = fmaxf((float)p.h2[1] * scale[ch + 1] + shift[ch + 1], 0.f);
  size_t o = (size_t)v * 32 + ch;
  float n0 = a0 + h[o], n1 = a1 + h[o + 1];
  h[o] = n0; h[o + 1] = n1;
  hf[o] = (_Float16)n0; hf[o + 1] = (_Float16)n1;
}

__global__ void pool_kernel(const float* __restrict__ h,
                            const int* __restrict__ batch,
                            float* __restrict__ sums, float* __restrict__ cnts) {
  int idx = blockIdx.x * blockDim.x + threadIdx.x;   // N*32 threads
  int v = idx >> 5, ch = idx & 31;
  int b = batch[v];
  atomicAdd(&sums[b * 32 + ch], h[idx]);
  if (ch == 0) atomicAdd(&cnts[b], 1.f);
}

__global__ void pred_kernel(const float* __restrict__ sums,
                            const float* __restrict__ cnts,
                            const float* __restrict__ pw,
                            const float* __restrict__ pb,
                            float* __restrict__ out) {
  int i = threadIdx.x;                                // 640 = 64*10
  int b = i / 10, o = i % 10;
  float c = fmaxf(cnts[b], 1.f);
  float acc = pb[o];
#pragma unroll
  for (int ch = 0; ch < 32; ++ch)
    acc += (sums[b * 32 + ch] / c) * pw[ch * 10 + o];
  out[b * 10 + o] = acc;
}

// ---------------------------------------------------------------------------
extern "C" void kernel_launch(void* const* d_in, const int* in_sizes, int n_in,
                              void* d_out, int out_size, void* d_ws, size_t ws_size,
                              hipStream_t stream) {
  const float* x       = (const float*)d_in[0];
  const int*   ei      = (const int*)d_in[1];
  const float* ea      = (const float*)d_in[2];
  const int*   batch   = (const int*)d_in[3];
  const float* lin_w   = (const float*)d_in[4];
  const float* lin_b   = (const float*)d_in[5];
  const float* msg_w1  = (const float*)d_in[6];
  const float* msg_b1  = (const float*)d_in[7];
  const float* msg_g1  = (const float*)d_in[8];
  const float* msg_be1 = (const float*)d_in[9];
  const float* msg_w2  = (const float*)d_in[10];
  const float* msg_b2  = (const float*)d_in[11];
  const float* msg_g2  = (const float*)d_in[12];
  const float* msg_be2 = (const float*)d_in[13];
  const float* upd_w1  = (const float*)d_in[14];
  const float* upd_b1  = (const float*)d_in[15];
  const float* upd_g1  = (const float*)d_in[16];
  const float* upd_be1 = (const float*)d_in[17];
  const float* upd_w2  = (const float*)d_in[18];
  const float* upd_b2  = (const float*)d_in[19];
  const float* upd_g2  = (const float*)d_in[20];
  const float* upd_be2 = (const float*)d_in[21];
  const float* pred_w  = (const float*)d_in[22];
  const float* pred_b  = (const float*)d_in[23];
  float* out = (float*)d_out;

  // Workspace layout (~138 MB, 256B-aligned blocks)
  char* W = (char*)d_ws;
  size_t off = 0;
  _Float16* wp  = (_Float16*)(W + off); off += (size_t)64 * 512 * 2;   // packed B frags
  float* gSum   = (float*)(W + off);    off += 256;                    // gSum[32]+gSqs[32]
  float* gSqs   = gSum + 32;
  float* scl    = (float*)(W + off);    off += 256;                    // scale[32]+shift[32]
  float* shf    = scl + 32;
  float* h      = (float*)(W + off);    off += (size_t)NN * 32 * 4;
  _Float16* hf  = (_Float16*)(W + off); off += (size_t)NN * 32 * 2;
  _Float16* eaf = (_Float16*)(W + off); off += (size_t)NE * 8 * 2;
  float* aggr   = (float*)(W + off);    off += (size_t)NN * 32 * 4;
  _Float16* m1f = (_Float16*)(W + off); off += (size_t)NE * 32 * 2;
  _Float16* m2  = (_Float16*)(W + off); off += (size_t)NE * 32 * 2;
  _Float16* u1f = (_Float16*)(W + off); off += (size_t)NN * 32 * 2;
  _Float16* u2  = (_Float16*)(W + off); off += (size_t)NN * 32 * 2;
  float* pools  = (float*)(W + off);    off += (size_t)NB * 32 * 4;
  float* cnts   = (float*)(W + off);    off += (size_t)NB * 4;

  // Pack all weights into f16 WMMA B fragments.
  pack_b_frags<<<8, 256, 0, stream>>>(lin_w, 128, wp);
  for (int l = 0; l < NL; ++l) {
    int b0 = 8 + l * 14;
    pack_b_frags<<<6, 256, 0, stream>>>(msg_w1 + l * 72 * 32, 72, wp + (size_t)b0 * 512);
    pack_b_frags<<<2, 256, 0, stream>>>(msg_w2 + l * 32 * 32, 32, wp + (size_t)(b0 + 6) * 512);
    pack_b_frags<<<4, 256, 0, stream>>>(upd_w1 + l * 64 * 32, 64, wp + (size_t)(b0 + 8) * 512);
    pack_b_frags<<<2, 256, 0, stream>>>(upd_w2 + l * 32 * 32, 32, wp + (size_t)(b0 + 12) * 512);
  }
  cvt_f16<<<25000, 256, 0, stream>>>(ea, eaf);           // E*8 elements exactly
  lin_in_gemm<<<625, 160, 0, stream>>>(x, wp, lin_b, h, hf);

  for (int l = 0; l < NL; ++l) {
    int b0 = 8 + l * 14;
    const _Float16* fw1 = wp + (size_t)b0 * 512;
    const _Float16* fw2 = wp + (size_t)(b0 + 6) * 512;
    const _Float16* fu1 = wp + (size_t)(b0 + 8) * 512;
    const _Float16* fu2 = wp + (size_t)(b0 + 12) * 512;

    hipMemsetAsync(gSum, 0, 256, stream);
    msg_gemm1<<<12500, 128, 0, stream>>>(hf, eaf, ei, fw1, msg_b1 + l * 32, m1f, gSum, gSqs);
    finalize_stats<<<1, 32, 0, stream>>>(gSum, gSqs, msg_g1 + l * 32, msg_be1 + l * 32,
                                         (float)NE, scl, shf);
    hipMemsetAsync(gSum, 0, 256, stream);
    gemm2_kernel<<<12500, 128, 0, stream>>>(m1f, fw2, scl, shf, msg_b2 + l * 32, m2, gSum, gSqs);
    finalize_stats<<<1, 32, 0, stream>>>(gSum, gSqs, msg_g2 + l * 32, msg_be2 + l * 32,
                                         (float)NE, scl, shf);
    hipMemsetAsync(aggr, 0, (size_t)NN * 32 * 4, stream);
    scatter_kernel<<<50000, 256, 0, stream>>>(m2, ei, scl, shf, aggr);

    hipMemsetAsync(gSum, 0, 256, stream);
    upd_gemm1<<<625, 160, 0, stream>>>(hf, aggr, fu1, upd_b1 + l * 32, u1f, gSum, gSqs);
    finalize_stats<<<1, 32, 0, stream>>>(gSum, gSqs, upd_g1 + l * 32, upd_be1 + l * 32,
                                         (float)NN, scl, shf);
    hipMemsetAsync(gSum, 0, 256, stream);
    gemm2_kernel<<<625, 160, 0, stream>>>(u1f, fu2, scl, shf, upd_b2 + l * 32, u2, gSum, gSqs);
    finalize_stats<<<1, 32, 0, stream>>>(gSum, gSqs, upd_g2 + l * 32, upd_be2 + l * 32,
                                         (float)NN, scl, shf);
    resid_kernel<<<3125, 256, 0, stream>>>(u2, scl, shf, h, hf);
  }

  hipMemsetAsync(pools, 0, (size_t)NB * 32 * 4 + NB * 4, stream);
  pool_kernel<<<6250, 256, 0, stream>>>(h, batch, pools, cnts);
  pred_kernel<<<1, 640, 0, stream>>>(pools, cnts, pred_w, pred_b, out);

  (void)in_sizes; (void)n_in; (void)out_size; (void)ws_size;
}
